// PointnetSAModuleMSGVotes_5669356832937
// MI455X (gfx1250) — compile-verified, matrix operand order checked by reading, not run on hardware
//
#include <hip/hip_runtime.h>

// ---------------------------------------------------------------------------
// PointNet++ SA-MSG module for gfx1250 (MI455X), wave32 + WMMA + TDM.
// ---------------------------------------------------------------------------

typedef __attribute__((ext_vector_type(16))) _Float16     v16h;
typedef __attribute__((ext_vector_type(8)))  float        v8f;
typedef __attribute__((ext_vector_type(4)))  unsigned int v4u;
typedef __attribute__((ext_vector_type(8)))  int          v8i;
typedef __attribute__((ext_vector_type(4)))  int          v4i;

constexpr int kB = 4;
constexpr int kN = 16384;
constexpr int kM = 1024;
constexpr int kC = 64;
constexpr int kQ = kB * kM;         // 4096 query points
constexpr int kKp0 = 96;            // layer-0 K padded 67 -> 96
constexpr float kEPS = 1e-5f;

#if __has_builtin(__builtin_amdgcn_tensor_load_to_lds)
#define USE_TDM 1
#if __has_include(<hip/amd_detail/amd_gfx1250_TDM.h>)
#define TDM_6ARG 1
#endif
#endif

// ---------------------------------------------------------------------------
// 1) Gather centroids + copy inds to output tail
// ---------------------------------------------------------------------------
__global__ void gather_kernel(const float* __restrict__ xyz,
                              const int* __restrict__ inds,
                              float* __restrict__ out_newxyz,
                              int* __restrict__ out_inds) {
  int t = blockIdx.x * blockDim.x + threadIdx.x;
  if (t >= kQ) return;
  int b = t / kM;
  int j = inds[t];
  const float* p = xyz + ((size_t)b * kN + j) * 3;
  out_newxyz[t * 3 + 0] = p[0];
  out_newxyz[t * 3 + 1] = p[1];
  out_newxyz[t * 3 + 2] = p[2];
  out_inds[t] = j;
}

// ---------------------------------------------------------------------------
// 2) Ball query: one wave32 per query; first `ns` indices in index order.
// ---------------------------------------------------------------------------
__global__ void ballquery_kernel(const float* __restrict__ xyz,
                                 const float* __restrict__ new_xyz,
                                 int* __restrict__ idx,
                                 float r2, int ns) {
  int lane = threadIdx.x & 31;
  int wave = (blockIdx.x * blockDim.x + threadIdx.x) >> 5;
  if (wave >= kQ) return;
  int b = wave / kM;
  float qx = new_xyz[wave * 3 + 0];
  float qy = new_xyz[wave * 3 + 1];
  float qz = new_xyz[wave * 3 + 2];
  const float* px = xyz + (size_t)b * kN * 3;
  int* row = idx + (size_t)wave * ns;

  int cnt = 0;
  for (int n0 = 0; n0 < kN && cnt < ns; n0 += 32) {
    int n = n0 + lane;
    float dx = px[n * 3 + 0] - qx;
    float dy = px[n * 3 + 1] - qy;
    float dz = px[n * 3 + 2] - qz;
    float d2 = dx * dx + dy * dy + dz * dz;
    unsigned mask = __builtin_amdgcn_ballot_w32(d2 < r2);
    if (lane == 0) {
      unsigned mm = mask;
      while (mm && cnt < ns) {
        int t = __builtin_ctz(mm);
        row[cnt++] = n0 + t;
        mm &= mm - 1;
      }
    }
    cnt = __builtin_amdgcn_readlane(cnt, 0);
  }
  if (lane == 0) {
    int f = (cnt > 0) ? row[0] : 0;
    for (int s = cnt; s < ns; ++s) row[s] = f;
  }
}

// ---------------------------------------------------------------------------
// 3) Group: build f16 A matrix (R x 96): [rel_xyz(3) | features(64) | pad(29)]
// ---------------------------------------------------------------------------
__global__ void group_kernel(const float* __restrict__ xyz,
                             const float* __restrict__ feats,
                             const float* __restrict__ new_xyz,
                             const int* __restrict__ idx,
                             _Float16* __restrict__ A,
                             int ns, int R) {
  int lane = threadIdx.x & 31;
  int r = (blockIdx.x * blockDim.x + threadIdx.x) >> 5;
  if (r >= R) return;
  int q = r / ns;            // query id (b,m)
  int b = q / kM;
  int j = idx[r];
  _Float16* row = A + (size_t)r * kKp0;
  if (lane < 3) {
    float v = xyz[((size_t)b * kN + j) * 3 + lane] - new_xyz[q * 3 + lane];
    row[lane] = (_Float16)v;
  }
  for (int c = lane; c < kC; c += 32) {
    row[3 + c] = (_Float16)feats[((size_t)b * kC + c) * kN + j];
  }
  for (int k = 3 + kC + lane; k < kKp0; k += 32) {
    row[k] = (_Float16)0.f;
  }
}

// ---------------------------------------------------------------------------
// 4a) Weight convert f32 (Co x Cin) -> f16 (Co x Kp), zero padded
// ---------------------------------------------------------------------------
__global__ void wconvert_kernel(const float* __restrict__ w,
                                _Float16* __restrict__ wh,
                                int co, int cin, int kp) {
  int t = blockIdx.x * blockDim.x + threadIdx.x;
  if (t >= co * kp) return;
  int o = t / kp, k = t % kp;
  wh[t] = (k < cin) ? (_Float16)w[(size_t)o * cin + k] : (_Float16)0.f;
}

// ---------------------------------------------------------------------------
// TDM: stage W tile (Co x K f16, row-major) from global into LDS.
// D# per ISA ch.8: group0 = {count, lds_addr, global_addr, type=2},
// group1 = {data_size=2B, tensor_dim0=K, tensor_dim1=Co, tile_dim0=K,
//           tile_dim1=Co, tensor_dim0_stride=K}; groups 2/3 zero (2-D tensor).
// ---------------------------------------------------------------------------
__device__ __forceinline__ void tdm_load_w(const _Float16* gptr,
                                           _Float16* sptr, int Co, int K) {
#ifdef USE_TDM
  unsigned lds = (unsigned)(size_t)sptr;
  unsigned long long ga = (unsigned long long)(size_t)gptr;
  v4u g0 = { 1u,                                   // count=1, gather off
             lds,                                  // lds_addr
             (unsigned)ga,                         // global_addr[31:0]
             ((unsigned)(ga >> 32) & 0x01FFFFFFu) | 0x80000000u }; // type=2
  v8i g1 = { (int)(1u << 16),                      // data_size = 2 bytes
             (int)((unsigned)K << 16),             // tensor_dim0[15:0]
             (int)((unsigned)Co << 16),            // tensor_dim1[15:0]
             (int)((unsigned)K << 16),             // tile_dim0
             Co,                                   // tile_dim1 (tile_dim2=0)
             K,                                    // tensor_dim0_stride[31:0]
             0, 0 };
  v4i gz = { 0, 0, 0, 0 };
#ifdef TDM_6ARG
  v8i gz8 = { 0, 0, 0, 0, 0, 0, 0, 0 };
  __builtin_amdgcn_tensor_load_to_lds(g0, g1, gz, gz, gz8, 0);
#else
  __builtin_amdgcn_tensor_load_to_lds(g0, g1, gz, gz, 0);
#endif
  __builtin_amdgcn_s_wait_tensorcnt(0);
#endif
}

// ---------------------------------------------------------------------------
// 4b) WMMA GEMM: Z (R x Co) f16 = A (R x K) f16 * W (Co x K)^T, f32 acc.
//     One wave per 16-row strip, NT = Co/16 accumulators; W tile in LDS.
//     Fragment layout per ISA 7.12.2: lane = 16*half + sub;
//       element j: k = kbase + (j/8)*16 + half*8 + j%8
//     -> per lane two contiguous 16B loads at (kbase + half*8) and (+16).
// ---------------------------------------------------------------------------
__device__ __forceinline__ v16h load_frag32(const _Float16* p) {
  union { uint4 u[2]; v16h h; } t;
  t.u[0] = *reinterpret_cast<const uint4*>(p);
  t.u[1] = *reinterpret_cast<const uint4*>(p + 16);
  return t.h;
}

template <int NT>
__global__ void wmma_gemm_kernel(const _Float16* __restrict__ A,
                                 const _Float16* __restrict__ W,
                                 _Float16* __restrict__ Z,
                                 int R, int K) {
  constexpr int Co = NT * 16;
  __shared__ _Float16 sW[128 * 96];     // 24 KB max (Co*K <= 128*96)

#ifdef USE_TDM
  if (threadIdx.x < 32) {               // wave 0 issues the TDM op + waits
    tdm_load_w(W, sW, Co, K);
  }
#else
  for (int t = threadIdx.x; t < Co * K; t += blockDim.x) sW[t] = W[t];
#endif
  __syncthreads();

  int lane = threadIdx.x & 31;
  int tm   = (blockIdx.x * blockDim.x + threadIdx.x) >> 5;  // 16-row strip
  if (tm >= (R >> 4)) return;          // wave-uniform: EXEC stays all-ones

  int hv  = lane >> 4;                 // lane half (0/1)
  int sub = lane & 15;                 // row within tile (A) / col (B)
  const _Float16* arow  = A  + (size_t)(tm * 16 + sub) * K + hv * 8;
  const _Float16* bbase = sW + (size_t)sub * K + hv * 8;

  v8f acc[NT];
#pragma unroll
  for (int t = 0; t < NT; ++t) acc[t] = (v8f){};

  for (int k = 0; k < K; k += 32) {
    __builtin_prefetch(arow + k + 64, 0, 1);
    v16h af = load_frag32(arow + k);
#pragma unroll
    for (int t = 0; t < NT; ++t) {
      v16h bf = load_frag32(bbase + (size_t)t * 16 * K + k);  // ds_load_b128
      acc[t] = __builtin_amdgcn_wmma_f32_16x16x32_f16(
          false, af, false, bf, (short)0, acc[t], false, false);
    }
  }

  // D layout: VGPR r -> M = tm*16 + r + half*8, N = tn*16 + sub
  int mbase = tm * 16 + hv * 8;
#pragma unroll
  for (int t = 0; t < NT; ++t) {
    int n = t * 16 + sub;
#pragma unroll
    for (int r = 0; r < 8; ++r) {
      Z[(size_t)(mbase + r) * Co + n] = (_Float16)acc[t][r];
    }
  }
}

// ---------------------------------------------------------------------------
// 4c) Per-channel column sum / sum-of-squares over Z (R x Co)
// ---------------------------------------------------------------------------
__global__ void colstats_kernel(const _Float16* __restrict__ Z,
                                float* __restrict__ sums,
                                float* __restrict__ sqs,
                                int R, int Co) {
  int c = blockIdx.x;
  int tid = threadIdx.x;
  float s = 0.f, q = 0.f;
  for (int r = tid; r < R; r += blockDim.x) {
    float v = (float)Z[(size_t)r * Co + c];
    s += v;
    q += v * v;
  }
  __shared__ float sh[512];
  sh[tid] = s;
  sh[256 + tid] = q;
  __syncthreads();
  for (int o = 128; o > 0; o >>= 1) {
    if (tid < o) {
      sh[tid] += sh[tid + o];
      sh[256 + tid] += sh[256 + tid + o];
    }
    __syncthreads();
  }
  if (tid == 0) {
    sums[c] = sh[0];
    sqs[c] = sh[256];
  }
}

// ---------------------------------------------------------------------------
// 4d) Fold BN stats + gamma/beta into per-channel scale/shift
// ---------------------------------------------------------------------------
__global__ void bnfinalize_kernel(const float* __restrict__ sums,
                                  const float* __restrict__ sqs,
                                  const float* __restrict__ g,
                                  const float* __restrict__ bta,
                                  float* __restrict__ gs,
                                  float* __restrict__ bs,
                                  int R, int Co) {
  int c = threadIdx.x;
  if (c >= Co) return;
  float invR = 1.f / (float)R;
  float mean = sums[c] * invR;
  float var = sqs[c] * invR - mean * mean;
  float s = g[c] * rsqrtf(var + kEPS);
  gs[c] = s;
  bs[c] = bta[c] - mean * s;
}

// ---------------------------------------------------------------------------
// 4e) affine + ReLU elementwise (intermediate layers): A = relu(gs*Z + bs)
// ---------------------------------------------------------------------------
__global__ void bnrelu_kernel(const _Float16* __restrict__ Z,
                              const float* __restrict__ gs,
                              const float* __restrict__ bs,
                              _Float16* __restrict__ A,
                              int total, int Co) {
  int t = blockIdx.x * blockDim.x + threadIdx.x;
  if (t >= total) return;
  int c = t % Co;
  float v = gs[c] * (float)Z[t] + bs[c];
  A[t] = (_Float16)fmaxf(v, 0.f);
}

// ---------------------------------------------------------------------------
// 4f) Final layer: affine + ReLU + max over ns -> (B, 256, M) slice
// ---------------------------------------------------------------------------
__global__ void bnrelu_maxpool_kernel(const _Float16* __restrict__ Z,
                                      const float* __restrict__ gs,
                                      const float* __restrict__ bs,
                                      float* __restrict__ out_feat,
                                      int ns, int coff) {
  int t = blockIdx.x * blockDim.x + threadIdx.x;
  if (t >= kQ * 128) return;
  int q = t >> 7;          // (b, m)
  int c = t & 127;
  float sc = gs[c], sh = bs[c];
  const _Float16* zp = Z + (size_t)q * ns * 128 + c;
  float mx = 0.f;          // post-ReLU values are >= 0
  for (int s = 0; s < ns; ++s) {
    float v = sc * (float)zp[(size_t)s * 128] + sh;
    v = fmaxf(v, 0.f);
    mx = fmaxf(mx, v);
  }
  int b = q / kM, m = q % kM;
  out_feat[((size_t)b * 256 + coff + c) * kM + m] = mx;
}

// ---------------------------------------------------------------------------
// Host orchestration
// ---------------------------------------------------------------------------
extern "C" void kernel_launch(void* const* d_in, const int* in_sizes, int n_in,
                              void* d_out, int out_size, void* d_ws, size_t ws_size,
                              hipStream_t stream) {
  (void)in_sizes; (void)n_in; (void)out_size; (void)ws_size;

  const float* xyz   = (const float*)d_in[0];
  const float* feats = (const float*)d_in[1];
  const int*   inds  = (const int*)d_in[2];

  float* out = (float*)d_out;
  float* out_newxyz = out;                                     // (B,M,3)
  float* out_feat   = out + (size_t)kQ * 3;                    // (B,256,M)
  int*   out_inds   = (int*)(out_feat + (size_t)kB * 256 * kM);// (B,M)

  // ---- carve workspace (sized for the larger scale, reused serially) ----
  size_t off = 0;
  auto carve = [&](size_t bytes) -> void* {
    void* p = (char*)d_ws + off;
    off += (bytes + 255) & ~(size_t)255;
    return p;
  };
  const size_t Rmax = (size_t)kQ * 32;                        // 131072 rows
  int*       idxbuf = (int*)carve(Rmax * sizeof(int));
  _Float16*  Abuf   = (_Float16*)carve(Rmax * 96  * sizeof(_Float16));
  _Float16*  Zbuf   = (_Float16*)carve(Rmax * 128 * sizeof(_Float16));
  _Float16*  Whbuf  = (_Float16*)carve((size_t)128 * 96 * sizeof(_Float16));
  float*     sums   = (float*)carve(128 * sizeof(float));
  float*     sqs    = (float*)carve(128 * sizeof(float));
  float*     gsbuf  = (float*)carve(128 * sizeof(float));
  float*     bsbuf  = (float*)carve(128 * sizeof(float));

  // ---- per-scale layer geometry (K padded to multiples of 32) ----
  const int NS[2]        = {16, 32};
  const float R2[2]      = {0.4f * 0.4f, 0.8f * 0.8f};
  const int Kdims[2][3]  = {{96, 64, 64}, {96, 64, 96}};   // padded K
  const int Cin[2][3]    = {{67, 64, 64}, {67, 64, 96}};   // true K
  const int Cdims[2][3]  = {{64, 64, 128}, {64, 96, 128}}; // Cout

  // 1) centroids + inds copy
  gather_kernel<<<(kQ + 255) / 256, 256, 0, stream>>>(xyz, inds, out_newxyz, out_inds);

  for (int sc = 0; sc < 2; ++sc) {
    const int ns = NS[sc];
    const int R  = kQ * ns;

    // 2) ball query: 4 waves / block
    ballquery_kernel<<<(kQ * 32 + 127) / 128, 128, 0, stream>>>(
        xyz, out_newxyz, idxbuf, R2[sc], ns);

    // 3) group into f16 A (R x 96): 8 waves / block
    group_kernel<<<((size_t)R * 32 + 255) / 256, 256, 0, stream>>>(
        xyz, feats, out_newxyz, idxbuf, Abuf, ns, R);

    for (int l = 0; l < 3; ++l) {
      const int K  = Kdims[sc][l];
      const int ci = Cin[sc][l];
      const int Co = Cdims[sc][l];
      const float* w   = (const float*)d_in[3 + sc * 9 + l * 3 + 0];
      const float* gam = (const float*)d_in[3 + sc * 9 + l * 3 + 1];
      const float* bet = (const float*)d_in[3 + sc * 9 + l * 3 + 2];

      // 4a) weights -> f16 padded
      wconvert_kernel<<<(Co * K + 255) / 256, 256, 0, stream>>>(w, Whbuf, Co, ci, K);

      // 4b) WMMA GEMM: one wave per 16-row strip, W staged via TDM into LDS
      int strips = R >> 4;
      dim3 ggrid((strips + 7) / 8);      // 8 waves / block
      switch (Co) {
        case 64:
          wmma_gemm_kernel<4><<<ggrid, 256, 0, stream>>>(Abuf, Whbuf, Zbuf, R, K);
          break;
        case 96:
          wmma_gemm_kernel<6><<<ggrid, 256, 0, stream>>>(Abuf, Whbuf, Zbuf, R, K);
          break;
        default:
          wmma_gemm_kernel<8><<<ggrid, 256, 0, stream>>>(Abuf, Whbuf, Zbuf, R, K);
          break;
      }

      // 4c/4d) BN stats
      colstats_kernel<<<Co, 256, 0, stream>>>(Zbuf, sums, sqs, R, Co);
      bnfinalize_kernel<<<1, 128, 0, stream>>>(sums, sqs, gam, bet, gsbuf, bsbuf, R, Co);

      if (l < 2) {
        // 4e) affine+ReLU -> next layer input
        bnrelu_kernel<<<((size_t)R * Co + 255) / 256, 256, 0, stream>>>(
            Zbuf, gsbuf, bsbuf, Abuf, R * Co, Co);
      } else {
        // 4f) affine+ReLU + max over ns -> output slice
        bnrelu_maxpool_kernel<<<(kQ * 128 + 255) / 256, 256, 0, stream>>>(
            Zbuf, gsbuf, bsbuf, out_feat, ns, sc * 128);
      }
    }
  }
}
